// WaveNet_66864050864721
// MI455X (gfx1250) — matrix operand order repaired and checked by reading, not compiled, not run on hardware
//
#include <hip/hip_runtime.h>
#include <hip/hip_bf16.h>
#include <math.h>

// ---------------------------------------------------------------------------
// WaveNet forward on MI455X (gfx1250), wave32 + V_WMMA_F32_16X16X32_BF16.
// All convs expressed as GEMMs: M = out-channels(256), N = time-tile(64),
// K = in-channels(256) x taps. bf16 operands, f32 accumulation.
// ---------------------------------------------------------------------------

typedef __attribute__((ext_vector_type(16))) __bf16 v16bf;
typedef __attribute__((ext_vector_type(8)))  __bf16 v8bf;
typedef __attribute__((ext_vector_type(8)))  float  v8f;

#define LL   4096      // sequence length
#define NB   32        // batch
#define CH   256       // channel width (RC=DC=SC=FC)
#define TL   64        // time tile per workgroup
#define ROWP 264       // padded LDS row (bf16 elems): 528B = 132 dwords
#define NTOK 30

// ---- fragment helpers ------------------------------------------------------

// 16 contiguous bf16 -> one operand fragment (two 16B loads).
__device__ __forceinline__ v16bf load_frag16(const __bf16* p) {
    const v8bf a = *(const v8bf*)p;
    const v8bf b = *(const v8bf*)(p + 8);
    return __builtin_shufflevector(a, b, 0,1,2,3,4,5,6,7,8,9,10,11,12,13,14,15);
}

__device__ __forceinline__ v8f wmma_bf16(v16bf a, v16bf b, v8f c) {
    // (neg_a, A, neg_b, B, c_mod, C, reuse_a, reuse_b)
    return __builtin_amdgcn_wmma_f32_16x16x32_bf16(false, a, false, b,
                                                   (short)0, c, false, false);
}

__device__ __forceinline__ v8f vzero8() {
    v8f z = {0.f,0.f,0.f,0.f,0.f,0.f,0.f,0.f};
    return z;
}

// ---------------------------------------------------------------------------
// Weight swizzle: fp32 W[m][k] (arbitrary strides) -> bf16 fragment layout.
// Element order used here (contiguous k, same as the LDS B fragments; 16x16x32 bf16):
//   lane = 16*half + (m & 15); element j -> k = k0 + 16*half + j
// Stored as [mt][kt][lane][j] contiguous so a lane's fragment = 32 contiguous B.
// ---------------------------------------------------------------------------
__global__ void wn_swizzleA(const float* __restrict__ src, __bf16* __restrict__ dst,
                            int Mdim, int Kdim, int strideM, int strideK, int Mvalid) {
    const int nkt = Kdim >> 5;
    const int total = Mdim * Kdim;
    for (int idx = blockIdx.x * blockDim.x + threadIdx.x; idx < total;
         idx += gridDim.x * blockDim.x) {
        const int j    = idx & 15;
        const int lane = (idx >> 4) & 31;
        const int tile = idx >> 9;
        const int kt   = tile % nkt;
        const int mt   = tile / nkt;
        const int hh   = lane >> 4;
        const int m    = mt * 16 + (lane & 15);
        const int k    = kt * 32 + 16 * hh + j;
        float v = 0.0f;
        if (m < Mvalid) v = src[(size_t)m * strideM + (size_t)k * strideK];
        dst[idx] = (__bf16)v;
    }
}

// ---------------------------------------------------------------------------
// Embedding gather + causal conv0 (EMB=128 -> 256, K=2, d=1).
// LDS tile: [TL+1][136] bf16 (1-element causal halo).
// ---------------------------------------------------------------------------
__global__ __launch_bounds__(256) void wn_embed0(
    const int* __restrict__ tokens, const float* __restrict__ table,
    const __bf16* __restrict__ W0A, const float* __restrict__ b0,
    float* __restrict__ resOut) {
    const int EMBP = 136;
    extern __shared__ __bf16 smE[];              // [TL+1][EMBP]
    const int b   = blockIdx.y;
    const int t0  = blockIdx.x * TL;
    const int tid = threadIdx.x;

    for (int i = tid; i < (TL + 1) * 8; i += 256) {
        const int t   = i >> 3;
        const int seg = (i & 7) * 16;
        const int gt  = t0 - 1 + t;
        __bf16* d = smE + t * EMBP + seg;
        if (gt < 0) {
            #pragma unroll
            for (int j = 0; j < 16; ++j) d[j] = (__bf16)0.0f;
        } else {
            const int tok = tokens[(size_t)b * LL + gt];
            const float* e = table + (size_t)tok * 128 + seg;
            #pragma unroll
            for (int j = 0; j < 16; ++j) d[j] = (__bf16)e[j];
        }
    }
    __syncthreads();

    const int wave = tid >> 5, lane = tid & 31;
    const int hh = lane >> 4, lx = lane & 15;

    v8f acc[2][4];
    #pragma unroll
    for (int mi = 0; mi < 2; ++mi)
        #pragma unroll
        for (int nt = 0; nt < 4; ++nt) acc[mi][nt] = vzero8();

    const size_t TAP0 = (size_t)CH * 128;        // per-tap swizzled size
    for (int kt = 0; kt < 4; ++kt) {
        const int kb = kt * 32 + 16 * hh;
        v16bf bp[4], bc[4];
        #pragma unroll
        for (int nt = 0; nt < 4; ++nt) {
            const int row = nt * 16 + lx;
            bp[nt] = load_frag16(smE + row * EMBP + kb);          // x[t-1]
            bc[nt] = load_frag16(smE + (row + 1) * EMBP + kb);    // x[t]
        }
        #pragma unroll
        for (int mi = 0; mi < 2; ++mi) {
            const int mt = wave * 2 + mi;
            const size_t ab = ((size_t)(mt * 4 + kt) * 32 + lane) * 16;
            const v16bf a0 = load_frag16(W0A + ab);
            const v16bf a1 = load_frag16(W0A + TAP0 + ab);
            #pragma unroll
            for (int nt = 0; nt < 4; ++nt) {
                acc[mi][nt] = wmma_bf16(a0, bp[nt], acc[mi][nt]);
                acc[mi][nt] = wmma_bf16(a1, bc[nt], acc[mi][nt]);
            }
        }
    }

    #pragma unroll
    for (int mi = 0; mi < 2; ++mi) {
        const int mt = wave * 2 + mi;
        #pragma unroll
        for (int r = 0; r < 8; ++r) {
            const int ch = mt * 16 + r + 8 * hh;
            const float bias = b0[ch];
            #pragma unroll
            for (int nt = 0; nt < 4; ++nt) {
                const int col = nt * 16 + lx;
                resOut[((size_t)b * CH + ch) * LL + t0 + col] = acc[mi][nt][r] + bias;
            }
        }
    }
}

// ---------------------------------------------------------------------------
// One residual block: z = tanh(Wf*res)*sigmoid(Wg*res) (dilated K=2 conv),
// skips (+)= Ws*z + bs ; resOut = resIn + Wr*z + br.
// ---------------------------------------------------------------------------
__global__ __launch_bounds__(256) void wn_block(
    const float* __restrict__ resIn, float* __restrict__ resOut,
    float* __restrict__ skips,
    const __bf16* __restrict__ WfA, const __bf16* __restrict__ WgA,
    const __bf16* __restrict__ WrA, const __bf16* __restrict__ WsA,
    const float* __restrict__ bfv, const float* __restrict__ bgv,
    const float* __restrict__ brv, const float* __restrict__ bsv,
    int dil, int first, int doRes) {
    extern __shared__ __bf16 smB[];
    __bf16* sPrev = smB;                         // [TL][ROWP] res[t-d]
    __bf16* sCur  = smB + TL * ROWP;             // [TL][ROWP] res[t]
    __bf16* sZ    = smB + 2 * TL * ROWP;         // [TL][ROWP] gated z

    const int b   = blockIdx.y;
    const int t0  = blockIdx.x * TL;
    const int tid = threadIdx.x;
    const float* rb = resIn + (size_t)b * CH * LL;

    // stage res tiles into LDS, transposed [time][channel], fp32 -> bf16
    for (int i = tid; i < CH * (TL / 4); i += 256) {
        const int ch = i >> 4;
        const int tq = (i & 15) * 4;
        const float4 c4 = *(const float4*)(rb + (size_t)ch * LL + t0 + tq);
        __bf16* dc = sCur + tq * ROWP + ch;
        dc[0]        = (__bf16)c4.x;
        dc[ROWP]     = (__bf16)c4.y;
        dc[2 * ROWP] = (__bf16)c4.z;
        dc[3 * ROWP] = (__bf16)c4.w;
        const int gp = t0 - dil + tq;
        __bf16* dp = sPrev + tq * ROWP + ch;
        #pragma unroll
        for (int e = 0; e < 4; ++e) {
            const int g = gp + e;
            dp[e * ROWP] = (__bf16)((g >= 0) ? rb[(size_t)ch * LL + g] : 0.0f);
        }
    }
    __syncthreads();

    const int wave = tid >> 5, lane = tid & 31;
    const int hh = lane >> 4, lx = lane & 15;
    const size_t TAP = (size_t)CH * CH;

    // ---- phase 1: gated dilated conv (two K=512 GEMMs sharing B operands)
    v8f accF[2][4], accG[2][4];
    #pragma unroll
    for (int mi = 0; mi < 2; ++mi)
        #pragma unroll
        for (int nt = 0; nt < 4; ++nt) { accF[mi][nt] = vzero8(); accG[mi][nt] = vzero8(); }

    for (int kt = 0; kt < 8; ++kt) {
        const int kb = kt * 32 + 16 * hh;
        v16bf bc[4], bp[4];
        #pragma unroll
        for (int nt = 0; nt < 4; ++nt) {
            const int row = nt * 16 + lx;
            bc[nt] = load_frag16(sCur + row * ROWP + kb);
            bp[nt] = load_frag16(sPrev + row * ROWP + kb);
        }
        #pragma unroll
        for (int mi = 0; mi < 2; ++mi) {
            const int mt = wave * 2 + mi;
            const size_t ab = ((size_t)(mt * 8 + kt) * 32 + lane) * 16;
            const v16bf af0 = load_frag16(WfA + ab);
            const v16bf af1 = load_frag16(WfA + TAP + ab);
            const v16bf ag0 = load_frag16(WgA + ab);
            const v16bf ag1 = load_frag16(WgA + TAP + ab);
            #pragma unroll
            for (int nt = 0; nt < 4; ++nt) {
                accF[mi][nt] = wmma_bf16(af0, bp[nt], accF[mi][nt]);
                accF[mi][nt] = wmma_bf16(af1, bc[nt], accF[mi][nt]);
                accG[mi][nt] = wmma_bf16(ag0, bp[nt], accG[mi][nt]);
                accG[mi][nt] = wmma_bf16(ag1, bc[nt], accG[mi][nt]);
            }
        }
    }

    // gated activation straight from C-layout accumulators -> z tile in LDS
    #pragma unroll
    for (int mi = 0; mi < 2; ++mi) {
        const int mt = wave * 2 + mi;
        #pragma unroll
        for (int r = 0; r < 8; ++r) {
            const int ch = mt * 16 + r + 8 * hh;
            const float fb = bfv[ch], gb = bgv[ch];
            #pragma unroll
            for (int nt = 0; nt < 4; ++nt) {
                const int col = nt * 16 + lx;
                const float zf = accF[mi][nt][r] + fb;
                const float zg = accG[mi][nt][r] + gb;
                const float z  = tanhf(zf) * (1.0f / (1.0f + __expf(-zg)));
                sZ[col * ROWP + ch] = (__bf16)z;
            }
        }
    }
    __syncthreads();

    // ---- phase 2: skip / residual 1x1 convs (K=256 GEMMs on z)
    v8f accS[2][4], accR[2][4];
    #pragma unroll
    for (int mi = 0; mi < 2; ++mi)
        #pragma unroll
        for (int nt = 0; nt < 4; ++nt) { accS[mi][nt] = vzero8(); accR[mi][nt] = vzero8(); }

    for (int kt = 0; kt < 8; ++kt) {
        const int kb = kt * 32 + 16 * hh;
        v16bf bz[4];
        #pragma unroll
        for (int nt = 0; nt < 4; ++nt)
            bz[nt] = load_frag16(sZ + (nt * 16 + lx) * ROWP + kb);
        #pragma unroll
        for (int mi = 0; mi < 2; ++mi) {
            const int mt = wave * 2 + mi;
            const size_t ab = ((size_t)(mt * 8 + kt) * 32 + lane) * 16;
            const v16bf as_ = load_frag16(WsA + ab);
            #pragma unroll
            for (int nt = 0; nt < 4; ++nt)
                accS[mi][nt] = wmma_bf16(as_, bz[nt], accS[mi][nt]);
            if (doRes) {
                const v16bf ar_ = load_frag16(WrA + ab);
                #pragma unroll
                for (int nt = 0; nt < 4; ++nt)
                    accR[mi][nt] = wmma_bf16(ar_, bz[nt], accR[mi][nt]);
            }
        }
    }

    // epilogue: skip accumulate + fp32 residual update
    #pragma unroll
    for (int mi = 0; mi < 2; ++mi) {
        const int mt = wave * 2 + mi;
        #pragma unroll
        for (int r = 0; r < 8; ++r) {
            const int ch = mt * 16 + r + 8 * hh;
            const float sb2 = bsv[ch], rb2 = brv[ch];
            #pragma unroll
            for (int nt = 0; nt < 4; ++nt) {
                const int col = nt * 16 + lx;
                const size_t gi = ((size_t)b * CH + ch) * LL + t0 + col;
                const float s = accS[mi][nt][r] + sb2;
                skips[gi] = first ? s : (skips[gi] + s);
                if (doRes) resOut[gi] = resIn[gi] + accR[mi][nt][r] + rb2;
            }
        }
    }
}

// ---------------------------------------------------------------------------
// Head: out = relu(skips); h = relu(Wfin*out + bfin); logits = Wout*h + bout;
// per-tile partial sum/max of h for the mean/max embeddings.
// ---------------------------------------------------------------------------
__global__ __launch_bounds__(256) void wn_head(
    const float* __restrict__ skips,
    const __bf16* __restrict__ WfinA, const float* __restrict__ bfin,
    const __bf16* __restrict__ WoutA, const float* __restrict__ bout,
    float* __restrict__ logits, float* __restrict__ psum, float* __restrict__ pmax) {
    extern __shared__ __bf16 smH[];
    __bf16* sS = smH;                            // relu(skips) tile
    __bf16* sH = smH + TL * ROWP;                // h tile

    const int b   = blockIdx.y;
    const int t0  = blockIdx.x * TL;
    const int tid = threadIdx.x;
    const float* sb = skips + (size_t)b * CH * LL;

    for (int i = tid; i < CH * (TL / 4); i += 256) {
        const int ch = i >> 4;
        const int tq = (i & 15) * 4;
        const float4 c4 = *(const float4*)(sb + (size_t)ch * LL + t0 + tq);
        __bf16* d = sS + tq * ROWP + ch;
        d[0]        = (__bf16)fmaxf(c4.x, 0.0f);
        d[ROWP]     = (__bf16)fmaxf(c4.y, 0.0f);
        d[2 * ROWP] = (__bf16)fmaxf(c4.z, 0.0f);
        d[3 * ROWP] = (__bf16)fmaxf(c4.w, 0.0f);
    }
    __syncthreads();

    const int wave = tid >> 5, lane = tid & 31;
    const int hh = lane >> 4, lx = lane & 15;

    // GEMM 1: h = relu(Wfin * relu(skips) + bfin)
    v8f acc[2][4];
    #pragma unroll
    for (int mi = 0; mi < 2; ++mi)
        #pragma unroll
        for (int nt = 0; nt < 4; ++nt) acc[mi][nt] = vzero8();

    for (int kt = 0; kt < 8; ++kt) {
        const int kb = kt * 32 + 16 * hh;
        v16bf bs2[4];
        #pragma unroll
        for (int nt = 0; nt < 4; ++nt)
            bs2[nt] = load_frag16(sS + (nt * 16 + lx) * ROWP + kb);
        #pragma unroll
        for (int mi = 0; mi < 2; ++mi) {
            const int mt = wave * 2 + mi;
            const v16bf a = load_frag16(WfinA + ((size_t)(mt * 8 + kt) * 32 + lane) * 16);
            #pragma unroll
            for (int nt = 0; nt < 4; ++nt)
                acc[mi][nt] = wmma_bf16(a, bs2[nt], acc[mi][nt]);
        }
    }

    // h to LDS + per-channel tile partials (sum / max over 64 time cols)
    #pragma unroll
    for (int mi = 0; mi < 2; ++mi) {
        const int mt = wave * 2 + mi;
        #pragma unroll
        for (int r = 0; r < 8; ++r) {
            const int ch = mt * 16 + r + 8 * hh;
            const float hb = bfin[ch];
            float hs = 0.0f, hm = 0.0f;          // h >= 0 so 0 is a safe max identity
            #pragma unroll
            for (int nt = 0; nt < 4; ++nt) {
                const int col = nt * 16 + lx;
                const float h = fmaxf(acc[mi][nt][r] + hb, 0.0f);
                sH[col * ROWP + ch] = (__bf16)h;
                hs += h;
                hm = fmaxf(hm, h);
            }
            #pragma unroll
            for (int m = 8; m > 0; m >>= 1) {    // butterfly within each 16-lane half
                hs += __shfl_xor(hs, m, 32);
                hm = fmaxf(hm, __shfl_xor(hm, m, 32));
            }
            if (lx == 0) {
                const size_t pi = ((size_t)b * (LL / TL) + blockIdx.x) * CH + ch;
                psum[pi] = hs;
                pmax[pi] = hm;
            }
        }
    }
    __syncthreads();

    // GEMM 2: logits (M padded to 32 -> 8 tiles, one per wave)
    const int mt2 = wave >> 2, nt2 = wave & 3;
    v8f accL = vzero8();
    for (int kt = 0; kt < 8; ++kt) {
        const v16bf a = load_frag16(WoutA + ((size_t)(mt2 * 8 + kt) * 32 + lane) * 16);
        const v16bf bb = load_frag16(sH + (nt2 * 16 + lx) * ROWP + kt * 32 + 16 * hh);
        accL = wmma_bf16(a, bb, accL);
    }
    #pragma unroll
    for (int r = 0; r < 8; ++r) {
        const int row = mt2 * 16 + r + 8 * hh;
        if (row < NTOK) {
            logits[((size_t)b * NTOK + row) * LL + t0 + nt2 * 16 + lx] = accL[r] + bout[row];
        }
    }
}

// Final mean / max over the 64 tile partials.
__global__ __launch_bounds__(256) void wn_reduce(
    const float* __restrict__ psum, const float* __restrict__ pmax,
    float* __restrict__ emb1, float* __restrict__ emb2) {
    const int b = blockIdx.x, ch = threadIdx.x;
    float s = 0.0f, m = 0.0f;
    for (int t = 0; t < LL / TL; ++t) {
        const size_t pi = ((size_t)b * (LL / TL) + t) * CH + ch;
        s += psum[pi];
        m = fmaxf(m, pmax[pi]);
    }
    emb1[(size_t)b * CH + ch] = s * (1.0f / LL);
    emb2[(size_t)b * CH + ch] = m;
}

// ---------------------------------------------------------------------------
extern "C" void kernel_launch(void* const* d_in, const int* in_sizes, int n_in,
                              void* d_out, int out_size, void* d_ws, size_t ws_size,
                              hipStream_t stream) {
    (void)in_sizes; (void)n_in; (void)out_size; (void)ws_size;

    const int*   tokens = (const int*)d_in[0];
    const float* table  = (const float*)d_in[1];
    const float* W0     = (const float*)d_in[2];
    const float* b0     = (const float*)d_in[3];
    const float* Wf     = (const float*)d_in[4];
    const float* bfp    = (const float*)d_in[5];
    const float* Wg     = (const float*)d_in[6];
    const float* bgp    = (const float*)d_in[7];
    const float* Wr     = (const float*)d_in[8];
    const float* brp    = (const float*)d_in[9];
    const float* Wsp    = (const float*)d_in[10];
    const float* bsp    = (const float*)d_in[11];
    const float* Wfin   = (const float*)d_in[12];
    const float* bfinp  = (const float*)d_in[13];
    const float* Wout   = (const float*)d_in[14];
    const float* boutp  = (const float*)d_in[15];

    char* ws = (char*)d_ws;
    const size_t SZ_RES = (size_t)NB * CH * LL * sizeof(float);     // 128 MB
    float* resA  = (float*)(ws);
    float* resB  = (float*)(ws + SZ_RES);
    float* skips = (float*)(ws + 2 * SZ_RES);
    __bf16* wb = (__bf16*)(ws + 3 * SZ_RES);

    size_t o = 0;
    __bf16* W0sw   = wb + o; o += (size_t)2 * CH * 128;
    __bf16* Wfsw   = wb + o; o += (size_t)16 * 2 * CH * CH;
    __bf16* Wgsw   = wb + o; o += (size_t)16 * 2 * CH * CH;
    __bf16* Wrsw   = wb + o; o += (size_t)16 * CH * CH;
    __bf16* Wssw   = wb + o; o += (size_t)16 * CH * CH;
    __bf16* Wfinsw = wb + o; o += (size_t)CH * CH;
    __bf16* Woutsw = wb + o; o += (size_t)32 * CH;
    float* psum = (float*)(wb + o);
    float* pmax = psum + (size_t)NB * (LL / TL) * CH;

    auto sw = [&](const float* src, __bf16* dst, int M, int K, int sM, int sK, int Mv) {
        const int total = M * K;
        wn_swizzleA<<<dim3((total + 255) / 256), dim3(256), 0, stream>>>(
            src, dst, M, K, sM, sK, Mv);
    };

    // ---- weight prep (fragment-swizzled bf16) ----
    for (int tap = 0; tap < 2; ++tap)
        sw(W0 + tap, W0sw + (size_t)tap * CH * 128, 256, 128, 128 * 2, 2, 256);
    for (int i = 0; i < 16; ++i) {
        for (int tap = 0; tap < 2; ++tap) {
            sw(Wf + (size_t)i * CH * CH * 2 + tap, Wfsw + ((size_t)i * 2 + tap) * CH * CH,
               256, 256, 512, 2, 256);
            sw(Wg + (size_t)i * CH * CH * 2 + tap, Wgsw + ((size_t)i * 2 + tap) * CH * CH,
               256, 256, 512, 2, 256);
        }
        sw(Wr + (size_t)i * CH * CH, Wrsw + (size_t)i * CH * CH, 256, 256, 256, 1, 256);
        sw(Wsp + (size_t)i * CH * CH, Wssw + (size_t)i * CH * CH, 256, 256, 256, 1, 256);
    }
    sw(Wfin, Wfinsw, 256, 256, 256, 1, 256);
    sw(Wout, Woutsw, 32, 256, 256, 1, NTOK);

    const dim3 grid(LL / TL, NB);

    // ---- embedding + conv0 ----
    wn_embed0<<<grid, 256, (size_t)(TL + 1) * 136 * sizeof(__bf16), stream>>>(
        tokens, table, W0sw, b0, resA);

    // ---- 16 residual blocks, res ping-pong ----
    const int dil[16] = {1,2,4,8,16,32,64,128, 1,2,4,8,16,32,64,128};
    float* rin = resA; float* rout = resB;
    const size_t shBlk = (size_t)3 * TL * ROWP * sizeof(__bf16);
    for (int i = 0; i < 16; ++i) {
        wn_block<<<grid, 256, shBlk, stream>>>(
            rin, rout, skips,
            Wfsw + (size_t)i * 2 * CH * CH, Wgsw + (size_t)i * 2 * CH * CH,
            Wrsw + (size_t)i * CH * CH, Wssw + (size_t)i * CH * CH,
            bfp + (size_t)i * CH, bgp + (size_t)i * CH,
            brp + (size_t)i * CH, bsp + (size_t)i * CH,
            dil[i], (i == 0) ? 1 : 0, (i == 15) ? 0 : 1);
        float* t = rin; rin = rout; rout = t;
    }

    // ---- head + embeddings ----
    float* logits = (float*)d_out;
    float* emb1 = logits + (size_t)NB * NTOK * LL;
    float* emb2 = emb1 + (size_t)NB * CH;
    wn_head<<<grid, 256, (size_t)2 * TL * ROWP * sizeof(__bf16), stream>>>(
        skips, Wfinsw, bfinp, Woutsw, boutp, logits, psum, pmax);
    wn_reduce<<<dim3(NB), dim3(CH), 0, stream>>>(psum, pmax, emb1, emb2);
}
